// MLPMixer_decoder_84146999263736
// MI455X (gfx1250) — matrix operand order repaired and checked by reading, not compiled, not run on hardware
//
#include <hip/hip_runtime.h>
#include <hip/hip_bf16.h>
#include <math.h>

// ---------------------------------------------------------------------------
// MI455X (gfx1250) implementation: all heavy GEMMs via v_wmma_f32_16x16x32_bf16
// Double-buffered LDS pipeline, branch-free (clamped) staging, L2 prefetch.
// ---------------------------------------------------------------------------

typedef __bf16 bf16;
typedef __attribute__((ext_vector_type(16))) __bf16 v16bf;
typedef __attribute__((ext_vector_type(8)))  __bf16 v8bf;
typedef __attribute__((ext_vector_type(8)))  float  v8f;

#define DEVI static __device__ __forceinline__

DEVI float gelu_exact(float x) { return 0.5f * x * (1.0f + erff(x * 0.7071067811865475f)); }
DEVI bf16  f2b(float x) { return (bf16)x; }
DEVI float b2f(bf16 x)  { return (float)x; }

// Problem constants
#define BB   16
#define NPT  1024      // N (tokens)
#define CHN  128
#define DMD  512
#define DFFN 1024
#define TOKD 256
#define N4   4096      // N * RATIO

// ------------------------------ WMMA GEMM ----------------------------------
// C[M,N] = act( alpha * A[M,K] * op(B) + bias ) + res
//   BT=1 : B stored [N,K] row-major (C = A * B^T)  -> Linear / scores
//   BT=0 : B stored [K,N] row-major (C = A * B)    -> attn-apply / conv layout
// Batched with per-operand batch strides; dual f32 + bf16 output (nullable).
#define MT 128
#define NTL 128
#define KT 32
#define LDSP 40   // padded LDS row stride (halfwords); 80B rows -> 16B aligned

template <bool BT>
__global__ __launch_bounds__(256) void gemm_bf16(
    const bf16* __restrict__ A, long long sA, int lda,
    const bf16* __restrict__ B, long long sB, int ldb,
    float* __restrict__ Cf, bf16* __restrict__ Cb, long long sC, int ldc,
    const float* __restrict__ bias, int bias_mode,   // 0 none, 1 per-n, 2 per-m
    const float* __restrict__ res, long long sR,
    int M, int N, int K, float alpha, int act)
{
    __shared__ __align__(16) bf16 As[2][MT * LDSP];
    __shared__ __align__(16) bf16 Bs[2][NTL * LDSP];

    const int z    = blockIdx.z;
    const bf16* Ab = A + (long long)z * sA;
    const bf16* Bb = B + (long long)z * sB;
    const int m0   = blockIdx.y * MT;
    const int n0   = blockIdx.x * NTL;
    const int tid  = threadIdx.x;
    const int lane = tid & 31;
    const int wave = tid >> 5;
    const int wm   = wave >> 2;   // 0..1 : 64 rows each
    const int wn   = wave & 3;    // 0..3 : 32 cols each
    const int fr   = lane & 15;
    const int half = lane >> 4;

    v8f acc[4][2];
#pragma unroll
    for (int i = 0; i < 4; ++i)
#pragma unroll
        for (int j = 0; j < 2; ++j)
#pragma unroll
            for (int e = 0; e < 8; ++e) acc[i][j][e] = 0.0f;

    // Per-thread staging registers (branch-free clamped loads; clamped rows are
    // never stored by the guarded epilogue, so duplicated data is harmless).
    unsigned long long ra[4], rb[4];

    auto ldA = [&](int k0) {
#pragma unroll
        for (int it = 0; it < 4; ++it) {
            int idx = tid + it * 256;          // 0..1023
            int r   = idx >> 3;                // 0..127
            int c   = (idx & 7) * 4;           // 0..28
            int gm  = m0 + r; gm = gm < M ? gm : M - 1;
            ra[it] = *(const unsigned long long*)(Ab + (long long)gm * lda + k0 + c);
        }
    };
    auto stA = [&](int nb) {
#pragma unroll
        for (int it = 0; it < 4; ++it) {
            int idx = tid + it * 256;
            int r   = idx >> 3;
            int c   = (idx & 7) * 4;
            *(unsigned long long*)(&As[nb][r * LDSP + c]) = ra[it];
        }
    };
    auto ldB = [&](int k0) {
        if (BT) {
#pragma unroll
            for (int it = 0; it < 4; ++it) {
                int idx = tid + it * 256;
                int r   = idx >> 3;
                int c   = (idx & 7) * 4;
                int gn  = n0 + r; gn = gn < N ? gn : N - 1;
                rb[it] = *(const unsigned long long*)(Bb + (long long)gn * ldb + k0 + c);
            }
        } else {
#pragma unroll
            for (int it = 0; it < 4; ++it) {
                int idx = tid + it * 256;
                int kk  = idx >> 5;            // 0..31
                int nl  = (idx & 31) * 4;      // 0..124
                int gn  = n0 + nl; gn = gn <= N - 4 ? gn : N - 4;
                rb[it] = *(const unsigned long long*)(Bb + (long long)(k0 + kk) * ldb + gn);
            }
        }
    };
    auto stB = [&](int nb) {
        if (BT) {
#pragma unroll
            for (int it = 0; it < 4; ++it) {
                int idx = tid + it * 256;
                int r   = idx >> 3;
                int c   = (idx & 7) * 4;
                *(unsigned long long*)(&Bs[nb][r * LDSP + c]) = rb[it];
            }
        } else {
#pragma unroll
            for (int it = 0; it < 4; ++it) {
                int idx = tid + it * 256;
                int kk  = idx >> 5;
                int nl  = (idx & 31) * 4;
                bf16 tmp[4];
                *(unsigned long long*)tmp = rb[it];
#pragma unroll
                for (int j = 0; j < 4; ++j) Bs[nb][(nl + j) * LDSP + kk] = tmp[j];
            }
        }
    };

    // prologue: stage tile 0
    ldA(0); ldB(0);
    stA(0); stB(0);
    __syncthreads();

    int cur = 0;
    for (int k0 = 0; k0 < K; k0 += KT) {
        const int nxt   = cur ^ 1;
        const bool more = (k0 + KT) < K;
        if (more) {
            ldA(k0 + KT);
            ldB(k0 + KT);
            if (k0 + 2 * KT < K) {  // pull tile-after-next toward L2/L0
                __builtin_prefetch(Ab + (long long)(m0 + (tid >> 3)) * lda + k0 + 2 * KT, 0, 3);
                __builtin_prefetch(Bb + (long long)(BT ? (n0 + (tid >> 3)) : (k0 + 2 * KT + (tid >> 5)))
                                            * ldb + (BT ? (k0 + 2 * KT) : n0), 0, 3);
            }
        }

        // ---- fragments from current LDS buffer ----
        v16bf afr[4];
#pragma unroll
        for (int i = 0; i < 4; ++i) {
            const bf16* p = &As[cur][(wm * 64 + i * 16 + fr) * LDSP];
            v8bf lo = *(const v8bf*)(p + half * 8);        // K = kbase..kbase+7
            v8bf hi = *(const v8bf*)(p + 16 + half * 8);   // K = 16+kbase..
            v16bf f;
#pragma unroll
            for (int e = 0; e < 8; ++e) { f[e] = lo[e]; f[e + 8] = hi[e]; }
            afr[i] = f;
        }
        v16bf bfr[2];
#pragma unroll
        for (int j = 0; j < 2; ++j) {
            const bf16* p = &Bs[cur][(wn * 32 + j * 16 + fr) * LDSP];
            v8bf lo = *(const v8bf*)(p + half * 8);        // K = half*8..+7
            v8bf hi = *(const v8bf*)(p + 16 + half * 8);   // K = 16+half*8..+7
            v16bf f;
#pragma unroll
            for (int e = 0; e < 8; ++e) { f[e] = lo[e]; f[e + 8] = hi[e]; }
            bfr[j] = f;
        }
#pragma unroll
        for (int i = 0; i < 4; ++i)
#pragma unroll
            for (int j = 0; j < 2; ++j)
                acc[i][j] = __builtin_amdgcn_wmma_f32_16x16x32_bf16(
                    false, afr[i], false, bfr[j], (short)0, acc[i][j], false, false);

        if (more) { stA(nxt); stB(nxt); }
        __syncthreads();
        cur = nxt;
    }

    // ---- epilogue ----
    float* Cfb       = Cf ? Cf + (long long)z * sC : nullptr;
    bf16*  Cbb       = Cb ? Cb + (long long)z * sC : nullptr;
    const float* Rb  = res ? res + (long long)z * sR : nullptr;
#pragma unroll
    for (int i = 0; i < 4; ++i) {
#pragma unroll
        for (int j = 0; j < 2; ++j) {
            int n = n0 + wn * 32 + j * 16 + fr;
#pragma unroll
            for (int r = 0; r < 8; ++r) {
                int m = m0 + wm * 64 + i * 16 + half * 8 + r;
                if (m < M && n < N) {
                    float v = alpha * acc[i][j][r];
                    if (bias_mode == 1) v += bias[n];
                    else if (bias_mode == 2) v += bias[m];
                    if (act) v = gelu_exact(v);
                    long long o = (long long)m * ldc + n;
                    if (Rb) v += Rb[o];
                    if (Cfb) Cfb[o] = v;
                    if (Cbb) Cbb[o] = f2b(v);
                }
            }
        }
    }
}

// --------------------------- transpose (tiled) -----------------------------
// out[z][c][r] = in[z][r][c] (+ add[z][c][r]); f32 and bf16 outputs nullable
__global__ __launch_bounds__(256) void xpose_k(
    const float* __restrict__ in, float* __restrict__ outf, bf16* __restrict__ outb,
    const float* __restrict__ add, int R, int Cc)
{
    __shared__ float t[32][33];
    int z = blockIdx.z;
    const float* ib = in + (long long)z * R * Cc;
    long long ob    = (long long)z * R * Cc;
    int r0 = blockIdx.x * 32, c0 = blockIdx.y * 32;
    int tx = threadIdx.x & 31, ty = threadIdx.x >> 5;
#pragma unroll
    for (int s = 0; s < 32; s += 8)
        t[ty + s][tx] = ib[(long long)(r0 + ty + s) * Cc + c0 + tx];
    __syncthreads();
#pragma unroll
    for (int s = 0; s < 32; s += 8) {
        int c = c0 + ty + s, r = r0 + tx;
        long long o = ob + (long long)c * R + r;
        float v = t[tx][ty + s];
        if (add) v += add[o];
        if (outf) outf[o] = v;
        if (outb) outb[o] = f2b(v);
    }
}

// ------------------------------ LayerNorm ----------------------------------
__global__ __launch_bounds__(128) void ln_k(
    const float* __restrict__ in, const float* __restrict__ g, const float* __restrict__ b,
    float* __restrict__ outf, bf16* __restrict__ outb, int W)
{
    int row = blockIdx.x;
    const float* x = in + (long long)row * W;
    __shared__ float red[128];
    float s = 0.0f;
    for (int i = threadIdx.x; i < W; i += 128) s += x[i];
    red[threadIdx.x] = s; __syncthreads();
    for (int o = 64; o > 0; o >>= 1) { if (threadIdx.x < o) red[threadIdx.x] += red[threadIdx.x + o]; __syncthreads(); }
    float mean = red[0] / (float)W; __syncthreads();
    float v = 0.0f;
    for (int i = threadIdx.x; i < W; i += 128) { float d = x[i] - mean; v += d * d; }
    red[threadIdx.x] = v; __syncthreads();
    for (int o = 64; o > 0; o >>= 1) { if (threadIdx.x < o) red[threadIdx.x] += red[threadIdx.x + o]; __syncthreads(); }
    float rstd = rsqrtf(red[0] / (float)W + 1e-5f);
    for (int i = threadIdx.x; i < W; i += 128) {
        float y = (x[i] - mean) * rstd * g[i] + b[i];
        long long o = (long long)row * W + i;
        if (outf) outf[o] = y;
        if (outb) outb[o] = f2b(y);
    }
}

// ------------------------- masked softmax (BigBird) ------------------------
__global__ __launch_bounds__(256) void softmax_k(
    float* __restrict__ sc, bf16* __restrict__ outb, const unsigned char* __restrict__ mask)
{
    int row = blockIdx.x;                 // b * NPT + n
    int n   = row & (NPT - 1);
    float* x = sc + (long long)row * NPT;
    const unsigned char* mk = mask + (long long)n * NPT;
    __shared__ float red[256];
    float vals[4];
    float mx = -3.0e38f;
#pragma unroll
    for (int i = 0; i < 4; ++i) {
        int c = threadIdx.x + i * 256;
        float v = mk[c] ? x[c] : -1.0e9f;
        vals[i] = v; mx = fmaxf(mx, v);
    }
    red[threadIdx.x] = mx; __syncthreads();
    for (int o = 128; o > 0; o >>= 1) { if (threadIdx.x < o) red[threadIdx.x] = fmaxf(red[threadIdx.x], red[threadIdx.x + o]); __syncthreads(); }
    mx = red[0]; __syncthreads();
    float s = 0.0f;
#pragma unroll
    for (int i = 0; i < 4; ++i) { vals[i] = __expf(vals[i] - mx); s += vals[i]; }
    red[threadIdx.x] = s; __syncthreads();
    for (int o = 128; o > 0; o >>= 1) { if (threadIdx.x < o) red[threadIdx.x] += red[threadIdx.x + o]; __syncthreads(); }
    float inv = 1.0f / red[0];
#pragma unroll
    for (int i = 0; i < 4; ++i) {
        int c = threadIdx.x + i * 256;
        float v = vals[i] * inv;
        x[c] = v;
        outb[(long long)row * NPT + c] = f2b(v);
    }
}

// ------------------------------ small kernels ------------------------------
__global__ __launch_bounds__(256) void cvt_k(const float* __restrict__ in, bf16* __restrict__ out, long long n)
{
    long long i = (long long)blockIdx.x * 256 + threadIdx.x;
    if (i < n) out[i] = f2b(in[i]);
}

// conv_x (3->64) + GELU, bf16 out [B,64,N]
__global__ __launch_bounds__(256) void convx_k(
    const float* __restrict__ coarse, const float* __restrict__ w, const float* __restrict__ b,
    bf16* __restrict__ out)
{
    long long idx = (long long)blockIdx.x * 256 + threadIdx.x;
    if (idx >= (long long)BB * 64 * NPT) return;
    int n  = (int)(idx & (NPT - 1));
    int o  = (int)((idx >> 10) & 63);
    int bb = (int)(idx >> 16);
    const float* x = coarse + (long long)bb * 3 * NPT + n;
    float s = b[o];
#pragma unroll
    for (int c = 0; c < 3; ++c) s += w[o * 3 + c] * x[(long long)c * NPT];
    out[idx] = f2b(gelu_exact(s));
}

// feat_g convs (N==1): 512->256 (+GELU), 256->128
__global__ __launch_bounds__(256) void fg1_k(const float* __restrict__ fg, const float* __restrict__ w,
                                             const float* __restrict__ b, float* __restrict__ out)
{
    int idx = blockIdx.x * 256 + threadIdx.x;
    if (idx >= BB * 256) return;
    int o = idx & 255, bb = idx >> 8;
    const float* x = fg + bb * 512;
    float s = b[o];
    for (int c = 0; c < 512; ++c) s += w[o * 512 + c] * x[c];
    out[idx] = gelu_exact(s);
}
__global__ __launch_bounds__(256) void fg2_k(const float* __restrict__ h, const float* __restrict__ w,
                                             const float* __restrict__ b, float* __restrict__ out)
{
    int idx = blockIdx.x * 256 + threadIdx.x;
    if (idx >= BB * 128) return;
    int o = idx & 127, bb = idx >> 7;
    const float* x = h + bb * 256;
    float s = b[o];
    for (int c = 0; c < 256; ++c) s += w[o * 256 + c] * x[c];
    out[idx] = s;
}

// concat(y, broadcast fg) -> row layout bf16 [B, N, 256]
__global__ __launch_bounds__(256) void buildx0_k(const float* __restrict__ y, const float* __restrict__ fg,
                                                 bf16* __restrict__ out)
{
    long long idx = (long long)blockIdx.x * 256 + threadIdx.x;
    if (idx >= (long long)BB * NPT * 256) return;
    int c  = (int)(idx & 255);
    int n  = (int)((idx >> 8) & (NPT - 1));
    int bb = (int)(idx >> 18);
    float v = (c < CHN) ? y[((long long)bb * CHN + c) * NPT + n] : fg[bb * CHN + (c - CHN)];
    out[idx] = f2b(v);
}

// y4 input gather: [B,256,4096] bf16 = concat(reshape(conv_ps out), tile(y,4))
__global__ __launch_bounds__(256) void gather_k(const float* __restrict__ ps, const float* __restrict__ y,
                                                bf16* __restrict__ out)
{
    long long idx = (long long)blockIdx.x * 256 + threadIdx.x;
    if (idx >= (long long)BB * 256 * N4) return;
    int n4 = (int)(idx & (N4 - 1));
    int c  = (int)((idx >> 12) & 255);
    int bb = (int)(idx >> 20);
    float v;
    if (c < CHN) {
        int c512 = c * 4 + (n4 >> 10);
        v = ps[((long long)bb * DMD + c512) * NPT + (n4 & (NPT - 1))];
    } else {
        v = y[((long long)bb * CHN + (c - CHN)) * NPT + (n4 & (NPT - 1))];
    }
    out[idx] = f2b(v);
}

// conv_out (64->3) + tiled-coarse residual -> d_out
__global__ __launch_bounds__(256) void convout_k(const bf16* __restrict__ t5, const float* __restrict__ w,
                                                 const float* __restrict__ b, const float* __restrict__ coarse,
                                                 float* __restrict__ out)
{
    long long idx = (long long)blockIdx.x * 256 + threadIdx.x;
    if (idx >= (long long)BB * 3 * N4) return;
    int n4 = (int)(idx & (N4 - 1));
    int o  = (int)((idx >> 12) % 3);
    int bb = (int)(idx / (3LL * N4));
    const bf16* x = t5 + (long long)bb * 64 * N4 + n4;
    float s = b[o];
    for (int c = 0; c < 64; ++c) s += w[o * 64 + c] * b2f(x[(long long)c * N4]);
    s += coarse[((long long)bb * 3 + o) * NPT + (n4 & (NPT - 1))];
    out[idx] = s;
}

// ------------------------------ host helpers -------------------------------
static void launch_gemm(hipStream_t st, bool bt,
                        const bf16* A, long long sA, int lda,
                        const bf16* B, long long sB, int ldb,
                        float* Cf, bf16* Cb, long long sC, int ldc,
                        const float* bias, int bias_mode,
                        const float* res, long long sR,
                        int M, int N, int K, float alpha, int act, int Z)
{
    dim3 grid((N + NTL - 1) / NTL, (M + MT - 1) / MT, Z);
    dim3 blk(256);
    if (bt)
        gemm_bf16<true><<<grid, blk, 0, st>>>(A, sA, lda, B, sB, ldb, Cf, Cb, sC, ldc,
                                              bias, bias_mode, res, sR, M, N, K, alpha, act);
    else
        gemm_bf16<false><<<grid, blk, 0, st>>>(A, sA, lda, B, sB, ldb, Cf, Cb, sC, ldc,
                                               bias, bias_mode, res, sR, M, N, K, alpha, act);
}

// Input index map (top-level dict order; params dict in insertion order)
enum {
    I_ASD = 0, I_COARSE, I_FEATG,
    I_CONVX_W, I_CONVX_B, I_CONVX1_W, I_CONVX1_B,
    I_CONV11_W, I_CONV11_B, I_CONV1_W, I_CONV1_B,
    I_CONVPS_W, I_CONVPS_B, I_CONVD_W, I_CONVD_B,
    I_CONVO1_W, I_CONVO1_B, I_CONVO_W, I_CONVO_B,
    I_SA1 = 19,  // pw,pb,n12g,n12b,n13g,n13b,w11,b11,w12,b12
    I_SA2 = 29,
    I_SA3 = 39,
    I_MB_LN1G = 49, I_MB_LN1B, I_MB_TW1, I_MB_TB1, I_MB_TW2, I_MB_TB2,
    I_MB_LN2G, I_MB_LN2B, I_MB_CW1, I_MB_CB1, I_MB_CW2, I_MB_CB2,
    I_MASK1 = 61, I_MASK2, I_MASK3
};

extern "C" void kernel_launch(void* const* d_in, const int* in_sizes, int n_in,
                              void* d_out, int out_size, void* d_ws, size_t ws_size,
                              hipStream_t stream)
{
    (void)in_sizes; (void)n_in; (void)out_size; (void)ws_size;
    const float* coarse = (const float*)d_in[I_COARSE];
    const float* featg  = (const float*)d_in[I_FEATG];
    float* out = (float*)d_out;
    auto F = [&](int i) { return (const float*)d_in[i]; };

    // ---- workspace carving ----
    char* wsp = (char*)d_ws;
    auto carve = [&](long long bytes) { void* q = wsp; wsp += (bytes + 255) & ~255LL; return q; };
    const long long F_Y = (long long)BB * CHN * NPT;       // 2M
    const long long F_ACT = (long long)BB * NPT * DMD;     // 8M
    const long long F_BIG = (long long)BB * NPT * NPT;     // 16M
    float* f_y   = (float*)carve(F_Y * 4);
    float* f_A   = (float*)carve(F_ACT * 4);
    float* f_B   = (float*)carve(F_ACT * 4);
    float* f_C   = (float*)carve(F_ACT * 4);
    float* f_big = (float*)carve(F_BIG * 4);
    float* f_fg1 = (float*)carve((long long)BB * 256 * 4);
    float* f_fg2 = (float*)carve((long long)BB * 128 * 4);
    bf16* h_A = (bf16*)carve(F_BIG * 2);
    bf16* h_B = (bf16*)carve(F_ACT * 2);
    bf16* h_C = (bf16*)carve(F_ACT * 2);
    bf16* h_D = (bf16*)carve(F_ACT * 2);
    bf16* h_W = (bf16*)carve(6LL * 1024 * 1024 * 2);

    // ---- weights -> bf16 ----
    bf16* wptr = h_W;
    auto wconv = [&](int idx, long long n) -> bf16* {
        bf16* dst = wptr; wptr += (n + 7) & ~7LL;
        cvt_k<<<dim3((unsigned)((n + 255) / 256)), dim3(256), 0, stream>>>(F(idx), dst, n);
        return dst;
    };
    bf16* Wx1   = wconv(I_CONVX1_W, 128 * 64);
    bf16* Wps   = wconv(I_CONVPS_W, 512 * 512);
    bf16* Wdel  = wconv(I_CONVD_W, 128 * 256);
    bf16* Wo1   = wconv(I_CONVO1_W, 64 * 128);
    bf16* Wtw1  = wconv(I_MB_TW1, TOKD * NPT);
    bf16* Wtw2  = wconv(I_MB_TW2, NPT * TOKD);
    bf16* Wcw1  = wconv(I_MB_CW1, 512 * 512);
    bf16* Wcw2  = wconv(I_MB_CW2, 512 * 512);
    bf16* Wproj[3], *W11[3], *W12[3];
    const int saBase[3] = { I_SA1, I_SA2, I_SA3 };
    const int saDin[3]  = { 256, 512, 512 };
    for (int s = 0; s < 3; ++s) {
        Wproj[s] = wconv(saBase[s] + 0, (long long)DMD * saDin[s]);
        W11[s]   = wconv(saBase[s] + 6, (long long)DFFN * DMD);
        W12[s]   = wconv(saBase[s] + 8, (long long)DMD * DFFN);
    }

    const int ROWS = BB * NPT;                   // 16384
    const long long S_ND = (long long)NPT * DMD; // 524288 per-batch stride
    const long long S_NN = (long long)NPT * NPT; // 1048576

    // ---- front end ----
    convx_k<<<dim3(4096), dim3(256), 0, stream>>>(coarse, F(I_CONVX_W), F(I_CONVX_B), h_B);
    // conv_x1: NN, A = W[128,64], B = [64,1024] per batch -> f_y [B,128,N]
    launch_gemm(stream, false, Wx1, 0, 64, h_B, 64LL * NPT, NPT,
                f_y, nullptr, (long long)CHN * NPT, NPT,
                F(I_CONVX1_B), 2, nullptr, 0, 128, NPT, 64, 1.0f, 0, BB);
    fg1_k<<<dim3(16), dim3(256), 0, stream>>>(featg, F(I_CONV11_W), F(I_CONV11_B), f_fg1);
    fg2_k<<<dim3(8),  dim3(256), 0, stream>>>(f_fg1, F(I_CONV1_W), F(I_CONV1_B), f_fg2);
    buildx0_k<<<dim3(16384), dim3(256), 0, stream>>>(f_y, f_fg2, h_B);  // [B,N,256] bf16

    // ---- three self-transformer blocks (row layout [B,N,512]) ----
    const int maskIdx[3] = { I_MASK1, I_MASK2, I_MASK3 };
    const float sc_alpha = 1.0f / sqrtf((float)DMD);
    for (int s = 0; s < 3; ++s) {
        int base = saBase[s], din = saDin[s];
        // P = X * Wproj^T + b    (flattened rows)
        launch_gemm(stream, true, h_B, 0, din, Wproj[s], 0, din,
                    f_C, nullptr, 0, DMD, F(base + 1), 1, nullptr, 0,
                    ROWS, DMD, din, 1.0f, 0, 1);
        // S = LN(P, n13)
        ln_k<<<dim3(ROWS), dim3(128), 0, stream>>>(f_C, F(base + 4), F(base + 5), f_A, h_C, DMD);
        // scores = S S^T * alpha   (per batch)
        launch_gemm(stream, true, h_C, S_ND, DMD, h_C, S_ND, DMD,
                    f_big, nullptr, S_NN, NPT, nullptr, 0, nullptr, 0,
                    NPT, NPT, DMD, sc_alpha, 0, BB);
        // masked softmax (writes f32 + bf16)
        softmax_k<<<dim3(ROWS), dim3(256), 0, stream>>>(f_big, h_A, (const unsigned char*)d_in[maskIdx[s]]);
        // X2 = S + A*S           (NN per batch, residual = S)
        launch_gemm(stream, false, h_A, S_NN, NPT, h_C, S_ND, DMD,
                    f_C, nullptr, S_ND, DMD, nullptr, 0, f_A, S_ND,
                    NPT, DMD, NPT, 1.0f, 0, BB);
        // T = LN(X2, n12)
        ln_k<<<dim3(ROWS), dim3(128), 0, stream>>>(f_C, F(base + 2), F(base + 3), f_B, h_B, DMD);
        // H = gelu(T W11^T + b11)   -> bf16 only
        launch_gemm(stream, true, h_B, 0, DMD, W11[s], 0, DMD,
                    nullptr, h_A, 0, DFFN, F(base + 7), 1, nullptr, 0,
                    ROWS, DFFN, DMD, 1.0f, 1, 1);
        // X3 = T + H W12^T + b12    -> f32 f_C + bf16 h_B (next block input)
        launch_gemm(stream, true, h_A, 0, DFFN, W12[s], 0, DFFN,
                    f_C, h_B, 0, DMD, F(base + 9), 1, f_B, 0,
                    ROWS, DMD, DFFN, 1.0f, 0, 1);
    }

    // ---- MixerBlock (x = f_C, [B,N,512]) ----
    ln_k<<<dim3(ROWS), dim3(128), 0, stream>>>(f_C, F(I_MB_LN1G), F(I_MB_LN1B), f_A, nullptr, DMD);
    // t' = transpose -> [B,512,N] bf16
    xpose_k<<<dim3(NPT / 32, DMD / 32, BB), dim3(256), 0, stream>>>(f_A, nullptr, h_C, nullptr, NPT, DMD);
    // u = gelu(t' tw1^T + tb1)  (M = B*512 flattened)
    launch_gemm(stream, true, h_C, 0, NPT, Wtw1, 0, NPT,
                nullptr, h_D, 0, TOKD, F(I_MB_TB1), 1, nullptr, 0,
                BB * DMD, TOKD, NPT, 1.0f, 1, 1);
    // u2 = u tw2^T + tb2 -> f_A [B,512,N]
    launch_gemm(stream, true, h_D, 0, TOKD, Wtw2, 0, TOKD,
                f_A, nullptr, 0, NPT, F(I_MB_TB2), 1, nullptr, 0,
                BB * DMD, NPT, TOKD, 1.0f, 0, 1);
    // x2 = x + u2^T -> f_B [B,N,512]
    xpose_k<<<dim3(DMD / 32, NPT / 32, BB), dim3(256), 0, stream>>>(f_A, f_B, nullptr, f_C, DMD, NPT);
    // c = LN(x2, ln2)
    ln_k<<<dim3(ROWS), dim3(128), 0, stream>>>(f_B, F(I_MB_LN2G), F(I_MB_LN2B), f_A, h_C, DMD);
    // h = gelu(c cw1^T + cb1)
    launch_gemm(stream, true, h_C, 0, DMD, Wcw1, 0, DMD,
                nullptr, h_D, 0, DMD, F(I_MB_CB1), 1, nullptr, 0,
                ROWS, DMD, DMD, 1.0f, 1, 1);
    // mixer_out = x2 + h cw2^T + cb2 -> f_C
    launch_gemm(stream, true, h_D, 0, DMD, Wcw2, 0, DMD,
                f_C, nullptr, 0, DMD, F(I_MB_CB2), 1, f_B, 0,
                ROWS, DMD, DMD, 1.0f, 0, 1);

    // ---- tail convs ----
    // conv layout: [B,N,512] -> [B,512,N] bf16
    xpose_k<<<dim3(NPT / 32, DMD / 32, BB), dim3(256), 0, stream>>>(f_C, nullptr, h_B, nullptr, NPT, DMD);
    // ps = conv_ps (NN, per batch) -> f_A [B,512,N]
    launch_gemm(stream, false, Wps, 0, DMD, h_B, (long long)DMD * NPT, NPT,
                f_A, nullptr, (long long)DMD * NPT, NPT,
                F(I_CONVPS_B), 2, nullptr, 0, DMD, NPT, DMD, 1.0f, 0, BB);
    // y4 input = concat(reshape(ps), tile(y)) bf16 [B,256,4096]
    gather_k<<<dim3(65536), dim3(256), 0, stream>>>(f_A, f_y, h_A);
    // y4 = conv_delta -> bf16 h_B [B,128,4096]
    launch_gemm(stream, false, Wdel, 0, 256, h_A, 256LL * N4, N4,
                nullptr, h_B, 128LL * N4, N4,
                F(I_CONVD_B), 2, nullptr, 0, 128, N4, 256, 1.0f, 0, BB);
    // t5 = gelu(conv_out1) -> bf16 h_C [B,64,4096]
    launch_gemm(stream, false, Wo1, 0, 128, h_B, 128LL * N4, N4,
                nullptr, h_C, 64LL * N4, N4,
                F(I_CONVO1_B), 2, nullptr, 0, 64, N4, 128, 1.0f, 1, BB);
    // x = conv_out + tile(coarse) -> d_out[0 : B*3*4096)
    convout_k<<<dim3(768), dim3(256), 0, stream>>>(h_C, F(I_CONVO_W), F(I_CONVO_B), coarse, out);
    // second tuple element: coarse passthrough
    hipMemcpyAsync(out + (long long)BB * 3 * N4, coarse,
                   (size_t)BB * 3 * NPT * sizeof(float), hipMemcpyDeviceToDevice, stream);
}
